// GRUAgent_53463752901212
// MI455X (gfx1250) — matrix operand order repaired and checked by reading, not compiled, not run on hardware
//
#include <hip/hip_runtime.h>
#include <math.h>

// ---------------- problem constants ----------------
#define TT   512
#define BB   1024
#define SS   64
#define AA   16
#define HH   128
#define H3   384

// ---------------- WMMA fragment types ----------------
typedef __attribute__((ext_vector_type(16))) __bf16 v16bf;
typedef __attribute__((ext_vector_type(8)))  float  v8f;
typedef __attribute__((ext_vector_type(8)))  unsigned short us8;
typedef int v4i __attribute__((__vector_size__(16)));

union AFrag {
  v16bf v;
  unsigned short u[16];
  us8 h[2];
};

// fp32 -> bf16 bits (round to nearest even)
__device__ __forceinline__ unsigned short f2bf(float f) {
  unsigned u = __float_as_uint(f);
  u += 0x7FFFu + ((u >> 16) & 1u);
  return (unsigned short)(u >> 16);
}

__device__ __forceinline__ float sigmoidf(float v) {
  return 1.0f / (1.0f + expf(-v));
}

// A-matrix 16x32 bf16 K pattern (ISA 7.12.2):
// lane<16:  v0..3 -> K 0..7, v4..7 -> K 16..23
// lane>=16: v0..3 -> K 8..15, v4..7 -> K 24..31
// => per lane, two contiguous 8-element runs: [8*hlf, +8) and [16+8*hlf, +8).
__device__ __forceinline__ int a_k_index(int j, int hlf) {
  return (j < 4 ? 2 * j : 16 + 2 * (j - 4)) + 8 * hlf;
}

// B-matrix 32x16 bf16: lane holds column n = lane&15; same K order as A:
// K [8*hlf, +8) then [16+8*hlf, +8). Column n of B == row (nbase+n) of W.
__device__ __forceinline__ void load_b(AFrag& B, const unsigned short* __restrict__ w,
                                       int ld, int nbase, int k0, int lane) {
  const int n = lane & 15;
  const int koff = k0 + ((lane >> 4) << 3);
  const unsigned short* p = w + (size_t)(nbase + n) * ld + koff;
#pragma unroll
  for (int i = 0; i < 16; ++i) B.u[i] = p[i < 8 ? i : i + 8];
}

__device__ __forceinline__ v8f wmma_bf16(const AFrag& a, const AFrag& b, v8f c) {
  return __builtin_amdgcn_wmma_f32_16x16x32_bf16(false, a.v, false, b.v,
                                                 (short)0, c, false, false);
}

// ---------------- gfx1250 async global->LDS copy (ASYNCcnt path) ----------------
#if __has_builtin(__builtin_amdgcn_global_load_async_to_lds_b128)
#define ASYNC_LDS 1
#endif

__device__ __forceinline__ void async_copy_b128(const float* gsrc, float* ldst) {
#ifdef ASYNC_LDS
  __builtin_amdgcn_global_load_async_to_lds_b128(
      (v4i*)gsrc,
      (__attribute__((address_space(3))) v4i*)ldst, 0, 0);
#else
  *(float4*)ldst = *(const float4*)gsrc;  // synchronous fallback
#endif
}

__device__ __forceinline__ void wait_async_all() {
#ifdef ASYNC_LDS
#if __has_builtin(__builtin_amdgcn_s_wait_asynccnt)
  __builtin_amdgcn_s_wait_asynccnt(0);
#else
  asm volatile("s_wait_asynccnt 0x0" ::: "memory");
#endif
#endif
}

// ---------------- kernel 0: convert weights to bf16 ----------------
__global__ __launch_bounds__(256) void convert_kernel(
    const float* __restrict__ W1, const float* __restrict__ W2,
    const float* __restrict__ Wih, const float* __restrict__ Whh,
    unsigned short* __restrict__ o1, unsigned short* __restrict__ o2,
    unsigned short* __restrict__ oih, unsigned short* __restrict__ ohh) {
  int i = blockIdx.x * 256 + threadIdx.x;
  if (i < HH * SS) o1[i] = f2bf(W1[i]);
  if (i < HH * HH) o2[i] = f2bf(W2[i]);
  if (i < H3 * HH) { oih[i] = f2bf(Wih[i]); ohh[i] = f2bf(Whh[i]); }
}

// ---------------- kernel 1: fused MLP encoder + input-gate GEMM ----------------
__global__ __launch_bounds__(256) void encoder_kernel(
    const float* __restrict__ x,
    const float* __restrict__ b1v, const float* __restrict__ b2v,
    const float* __restrict__ bihv,
    const unsigned short* __restrict__ W1b,
    const unsigned short* __restrict__ W2b,
    const unsigned short* __restrict__ Wihb,
    float* __restrict__ gx) {
  __shared__ __attribute__((aligned(16))) unsigned short stage[8][16 * HH];  // 32 KB
  const int wave = threadIdx.x >> 5;
  const int lane = threadIdx.x & 31;
  const int m = lane & 15;
  const int hlf = lane >> 4;
  const long row0 = (long)blockIdx.x * 128 + (long)wave * 16;
  unsigned short* st = stage[wave];

  // ---- layer 1: [16,64] @ W1^T -> [16,128]
  AFrag xa[2];
  {
    const float* xr = x + (row0 + m) * SS;
#pragma unroll
    for (int f = 0; f < 2; ++f)
#pragma unroll
      for (int j = 0; j < 8; ++j) {
        int k = a_k_index(j, hlf) + 32 * f;
        xa[f].u[2 * j]     = f2bf(xr[k]);
        xa[f].u[2 * j + 1] = f2bf(xr[k + 1]);
      }
  }
#pragma unroll
  for (int nt = 0; nt < 8; ++nt) {
    v8f acc = {};
#pragma unroll
    for (int f = 0; f < 2; ++f) {
      AFrag bf;
      load_b(bf, W1b, SS, nt * 16, 32 * f, lane);
      acc = wmma_bf16(xa[f], bf, acc);
    }
    const float bias = b1v[nt * 16 + m];
#pragma unroll
    for (int r = 0; r < 8; ++r)
      st[(r + 8 * hlf) * HH + nt * 16 + m] = f2bf(fmaxf(acc[r] + bias, 0.0f));
  }
  __syncthreads();

  // ---- layer 2: [16,128] @ W2^T -> [16,128]
  AFrag ha[4];
#pragma unroll
  for (int f = 0; f < 4; ++f) {
    ha[f].h[0] = *(const us8*)&st[m * HH + 32 * f + 8 * hlf];
    ha[f].h[1] = *(const us8*)&st[m * HH + 32 * f + 8 * hlf + 16];
  }
  __syncthreads();
#pragma unroll
  for (int nt = 0; nt < 8; ++nt) {
    v8f acc = {};
#pragma unroll
    for (int f = 0; f < 4; ++f) {
      AFrag bf;
      load_b(bf, W2b, HH, nt * 16, 32 * f, lane);
      acc = wmma_bf16(ha[f], bf, acc);
    }
    const float bias = b2v[nt * 16 + m];
#pragma unroll
    for (int r = 0; r < 8; ++r)
      st[(r + 8 * hlf) * HH + nt * 16 + m] = f2bf(fmaxf(acc[r] + bias, 0.0f));
  }
  __syncthreads();

  // ---- layer 3: gx = h2 @ W_ih^T + b_ih  [16,384]
  AFrag h2a[4];
#pragma unroll
  for (int f = 0; f < 4; ++f) {
    h2a[f].h[0] = *(const us8*)&st[m * HH + 32 * f + 8 * hlf];
    h2a[f].h[1] = *(const us8*)&st[m * HH + 32 * f + 8 * hlf + 16];
  }
#pragma unroll
  for (int nt = 0; nt < 24; ++nt) {
    v8f acc = {};
#pragma unroll
    for (int f = 0; f < 4; ++f) {
      AFrag bf;
      load_b(bf, Wihb, HH, nt * 16, 32 * f, lane);
      acc = wmma_bf16(h2a[f], bf, acc);
    }
    const float bias = bihv[nt * 16 + m];
#pragma unroll
    for (int r = 0; r < 8; ++r)
      gx[(row0 + r + 8 * hlf) * H3 + nt * 16 + m] = acc[r] + bias;
  }
}

// ---------------- kernel 2: sequential GRU scan ----------------
// 64 blocks x 16 batch rows. h lives in LDS twice: fp32 (exact state) and a
// bf16 mirror (WMMA A-operand, loaded as aligned ds b128). gx tiles are
// double-buffered in LDS via async global->LDS b128 copies (ASYNCcnt).
__global__ __launch_bounds__(256) void gru_scan_kernel(
    const float* __restrict__ gx,    // [T,B,384]
    const float* __restrict__ done,  // [T*B]
    const float* __restrict__ h0,    // [B,H]
    const float* __restrict__ bhh,   // [384]
    const unsigned short* __restrict__ Whhb,  // [384,128] bf16
    float* __restrict__ hidden,      // [T*B,H]
    float* __restrict__ hlast) {     // [B,H]
  __shared__ __attribute__((aligned(16))) float gxbuf[2][16 * H3];        // 48 KB
  __shared__ __attribute__((aligned(16))) float hbuf32[16 * HH];          // 8 KB
  __shared__ __attribute__((aligned(16))) unsigned short hbufb[16 * HH];  // 4 KB
  const int tid = threadIdx.x;
  const int wave = tid >> 5;
  const int lane = tid & 31;
  const int m = lane & 15;
  const int hlf = lane >> 4;
  const int b0 = blockIdx.x * 16;
  const int col = wave * 16 + m;

  for (int i = tid; i < 16 * HH; i += 256) {
    float v = h0[(long)b0 * HH + i];
    hbuf32[i] = v;
    hbufb[i] = f2bf(v);
  }

  AFrag Bf[3][4];
#pragma unroll
  for (int g = 0; g < 3; ++g)
#pragma unroll
    for (int f = 0; f < 4; ++f)
      load_b(Bf[g][f], Whhb, HH, g * HH + wave * 16, 32 * f, lane);

  const float br = bhh[col], bz = bhh[HH + col], bn = bhh[2 * HH + col];

  // preload gx tile for t=0 (tile is contiguous 16*384 floats)
  {
    const float* src = gx + (long)b0 * H3;
#pragma unroll
    for (int q = 0; q < 6; ++q)
      async_copy_b128(src + (q * 256 + tid) * 4, &gxbuf[0][(q * 256 + tid) * 4]);
  }
  wait_async_all();
  __syncthreads();

  for (int t = 0; t < TT; ++t) {
    const long tb = (long)t * BB + b0;
    const int cur = t & 1;

    // kick off async copy of next step's gx tile into the other buffer
    if (t + 1 < TT) {
      const float* src = gx + (tb + BB) * H3;
      float* dst = gxbuf[cur ^ 1];
#pragma unroll
      for (int q = 0; q < 6; ++q)
        async_copy_b128(src + (q * 256 + tid) * 4, dst + (q * 256 + tid) * 4);
      if (t + 2 < TT && tid < 96)
        __builtin_prefetch(gx + (tb + 2 * BB) * H3 + tid * 64, 0, 3);
    }

    // done-mask reset of h (done is exactly 0.0/1.0)
    for (int i = tid; i < 16 * HH; i += 256) {
      float s = 1.0f - done[tb + (i >> 7)];
      hbuf32[i] *= s;
      if (s == 0.0f) hbufb[i] = 0;
    }
    __syncthreads();

    // A fragments straight from the bf16 mirror: 2 aligned b128 LDS loads each
    AFrag Af[4];
#pragma unroll
    for (int f = 0; f < 4; ++f) {
      Af[f].h[0] = *(const us8*)&hbufb[m * HH + 32 * f + 8 * hlf];
      Af[f].h[1] = *(const us8*)&hbufb[m * HH + 32 * f + 8 * hlf + 16];
    }
    float hc[8];
#pragma unroll
    for (int r = 0; r < 8; ++r) hc[r] = hbuf32[(r + 8 * hlf) * HH + col];
    __syncthreads();  // all reads done before anyone writes h_new

    v8f ar = {}, az = {}, an = {};
#pragma unroll
    for (int f = 0; f < 4; ++f) {
      ar = wmma_bf16(Af[f], Bf[0][f], ar);
      az = wmma_bf16(Af[f], Bf[1][f], az);
      an = wmma_bf16(Af[f], Bf[2][f], an);
    }

#pragma unroll
    for (int r = 0; r < 8; ++r) {
      const int lrow = r + 8 * hlf;
      const float* gp = &gxbuf[cur][lrow * H3];
      float rr = sigmoidf(gp[col] + ar[r] + br);
      float zz = sigmoidf(gp[HH + col] + az[r] + bz);
      float nn = tanhf(gp[2 * HH + col] + rr * (an[r] + bn));
      float hv = (1.0f - zz) * nn + zz * hc[r];
      hbuf32[lrow * HH + col] = hv;
      hbufb[lrow * HH + col] = f2bf(hv);
      hidden[(tb + lrow) * HH + col] = hv;
    }
    wait_async_all();  // next tile fully landed in LDS
    __syncthreads();
  }

  for (int i = tid; i < 16 * HH; i += 256)
    hlast[(long)b0 * HH + i] = hbuf32[i];
}

// ---------------- kernel 3: policy/value heads ----------------
__global__ __launch_bounds__(256) void heads_kernel(
    const float* __restrict__ hidden, const int* __restrict__ action,
    const float* __restrict__ Wa, const float* __restrict__ ba,
    const float* __restrict__ Wc, const float* __restrict__ bc,
    float* __restrict__ outm) {
  const int wave = threadIdx.x >> 5;
  const int lane = threadIdx.x & 31;
  const long row = (long)blockIdx.x * 8 + wave;

  const float4 h4 = *(const float4*)(hidden + row * HH + lane * 4);
  float acc[17];
#pragma unroll
  for (int n = 0; n < AA; ++n) {
    const float* w = Wa + n * HH + lane * 4;
    acc[n] = h4.x * w[0] + h4.y * w[1] + h4.z * w[2] + h4.w * w[3];
  }
  {
    const float* w = Wc + lane * 4;
    acc[16] = h4.x * w[0] + h4.y * w[1] + h4.z * w[2] + h4.w * w[3];
  }
#pragma unroll
  for (int off = 16; off > 0; off >>= 1)
#pragma unroll
    for (int n = 0; n < 17; ++n) acc[n] += __shfl_xor(acc[n], off, 32);

  float lg[AA], mx = -INFINITY;
#pragma unroll
  for (int n = 0; n < AA; ++n) {
    lg[n] = acc[n] + ba[n];
    mx = fmaxf(mx, lg[n]);
  }
  float se = 0.0f;
#pragma unroll
  for (int n = 0; n < AA; ++n) se += expf(lg[n] - mx);
  const float lse = mx + logf(se);

  float ent = 0.0f;
#pragma unroll
  for (int n = 0; n < AA; ++n) {
    float lp = lg[n] - lse;
    ent -= expf(lp) * lp;
  }
  if (lane == 0) {
    const int a = action[row];
    outm[row * 3 + 0] = lg[a] - lse;
    outm[row * 3 + 1] = ent;
    outm[row * 3 + 2] = acc[16] + bc[0];
  }
}

// ---------------- launcher ----------------
extern "C" void kernel_launch(void* const* d_in, const int* in_sizes, int n_in,
                              void* d_out, int out_size, void* d_ws, size_t ws_size,
                              hipStream_t stream) {
  const float* x    = (const float*)d_in[0];
  const float* done = (const float*)d_in[1];
  const int* action = (const int*)d_in[2];
  const float* h0   = (const float*)d_in[3];
  const float* W1   = (const float*)d_in[4];
  const float* b1   = (const float*)d_in[5];
  const float* W2   = (const float*)d_in[6];
  const float* b2   = (const float*)d_in[7];
  const float* Wih  = (const float*)d_in[8];
  const float* bih  = (const float*)d_in[9];
  const float* Whh  = (const float*)d_in[10];
  const float* bhh  = (const float*)d_in[11];
  const float* Wa   = (const float*)d_in[12];
  const float* ba   = (const float*)d_in[13];
  const float* Wc   = (const float*)d_in[14];
  const float* bc   = (const float*)d_in[15];
  float* out = (float*)d_out;

  char* ws = (char*)d_ws;
  size_t off = 0;
  float* gx = (float*)(ws + off);              off += (size_t)TT * BB * H3 * 4;
  float* hidden = (float*)(ws + off);          off += (size_t)TT * BB * HH * 4;
  unsigned short* W1b = (unsigned short*)(ws + off);  off += (size_t)HH * SS * 2;
  unsigned short* W2b = (unsigned short*)(ws + off);  off += (size_t)HH * HH * 2;
  unsigned short* Wihb = (unsigned short*)(ws + off); off += (size_t)H3 * HH * 2;
  unsigned short* Whhb = (unsigned short*)(ws + off); off += (size_t)H3 * HH * 2;

  hipLaunchKernelGGL(convert_kernel, dim3((H3 * HH + 255) / 256), dim3(256), 0, stream,
                     W1, W2, Wih, Whh, W1b, W2b, Wihb, Whhb);
  hipLaunchKernelGGL(encoder_kernel, dim3((TT * BB) / 128), dim3(256), 0, stream,
                     x, b1, b2, bih, W1b, W2b, Wihb, gx);
  hipLaunchKernelGGL(gru_scan_kernel, dim3(BB / 16), dim3(256), 0, stream,
                     gx, done, h0, bhh, Whhb, hidden, out + (size_t)TT * BB * 3);
  hipLaunchKernelGGL(heads_kernel, dim3((TT * BB) / 8), dim3(256), 0, stream,
                     hidden, action, Wa, ba, Wc, bc, out);
}
